// HAN_65807488909778
// MI455X (gfx1250) — compile-verified, matrix operand order checked by reading, not run on hardware
//
#include <hip/hip_runtime.h>
#include <hip/hip_bf16.h>

// ---------------------------------------------------------------------------
// Types for CDNA5 WMMA bf16 path
// ---------------------------------------------------------------------------
typedef __attribute__((ext_vector_type(16))) __bf16 v16bf;
typedef __attribute__((ext_vector_type(8)))  __bf16 v8bf;
typedef __attribute__((ext_vector_type(8)))  float  v8f;

union FragU { v16bf f; v8bf h[2]; };

#define WMMA_BF16(A, B, C) \
  __builtin_amdgcn_wmma_f32_16x16x32_bf16(false, (A), false, (B), (short)0, (C), false, false)

// Load a 16(rows) x 32(K) bf16 fragment from row-major [rows][ld] storage.
// ISA 16-bit A layout: lanes 0-15 -> K[0..7] in v0..3, K[16..23] in v4..7;
// lanes 16-31 -> K[8..15], K[24..31]. B fragments use the same pattern on the
// N x K (row-major) transposed operand.
__device__ inline v16bf load_frag_bf16(const __bf16* __restrict__ base, int ld,
                                       int row0, int k0) {
  int lane = threadIdx.x & 31;
  int half = lane >> 4;
  int r    = lane & 15;
  const __bf16* p = base + (size_t)(row0 + r) * ld + k0 + half * 8;
  FragU u;
  u.h[0] = *(const v8bf*)(p);        // K = half*8 .. half*8+7
  u.h[1] = *(const v8bf*)(p + 16);   // K = 16+half*8 .. +7
  return u.f;
}

// ---------------------------------------------------------------------------
// CDNA5 async global->LDS copy (16 bytes per lane), tracked by ASYNCcnt.
// LDS operand is the low 32 bits of the flat address (ISA aperture mapping).
// ---------------------------------------------------------------------------
__device__ inline void async_copy_b128(void* lds_dst, const void* gsrc) {
  unsigned l = (unsigned)(uintptr_t)lds_dst;
  asm volatile("global_load_async_to_lds_b128 %0, %1, off"
               :: "v"(l), "v"(gsrc) : "memory");
}
__device__ inline void wait_asynccnt0() {
  asm volatile("s_wait_asynccnt 0" ::: "memory");
}

// ---------------------------------------------------------------------------
// Problem constants
// ---------------------------------------------------------------------------
#define BB    128
#define NQ    16
#define NS    256
#define EDIM  300
#define KPACK 928      // 900 padded to multiple of 32
#define CDIM  1024
#define HH    8
#define NOUT  300
#define NPADO 320      // 300 padded to mult of 32 (K for glove gemm)
#define FCNP  512      // fc output N padded to block tile (256) multiple
#define NANS  5000
#define NAPAD 5120     // 5000 padded to 256 multiple
#define FCK   8192     // H*C

// ---------------------------------------------------------------------------
// Gather emb rows for 3 nodes per hyperedge -> bf16 packed [rows][KPACK]
// ---------------------------------------------------------------------------
__global__ __launch_bounds__(256)
void pack_rows_kernel(const int* __restrict__ idx, const float* __restrict__ emb,
                      __bf16* __restrict__ out) {
  int r = blockIdx.x;
  int i0 = idx[r * 3 + 0];
  int i1 = idx[r * 3 + 1];
  int i2 = idx[r * 3 + 2];
  for (int k = threadIdx.x; k < KPACK; k += 256) {
    float v = 0.0f;
    if (k < 300)      v = emb[(size_t)i0 * EDIM + k];
    else if (k < 600) v = emb[(size_t)i1 * EDIM + (k - 300)];
    else if (k < 900) v = emb[(size_t)i2 * EDIM + (k - 600)];
    out[(size_t)r * KPACK + k] = (__bf16)v;
  }
}

// f32 [Rsrc][Csrc] -> bf16 [gridDim.x][Cdst], zero padded rows/cols
__global__ __launch_bounds__(256)
void cvt_pad_kernel(const float* __restrict__ src, int Rsrc, int Csrc,
                    __bf16* __restrict__ dst, int Cdst) {
  int r = blockIdx.x;
  for (int k = threadIdx.x; k < Cdst; k += 256) {
    float v = (r < Rsrc && k < Csrc) ? src[(size_t)r * Csrc + k] : 0.0f;
    dst[(size_t)r * Cdst + k] = (__bf16)v;
  }
}

// ---------------------------------------------------------------------------
// Generic WMMA GEMM: out[m][n] = sum_k A[m][k] * W[n][k] + bias[n]
// 256 threads = 8 waves arranged 2(M) x 4(N); each wave computes a 64x64
// macro-tile (4x4 WMMA tiles) -> block tile 128M x 256N.
// 16 b128 loads : 16 WMMAs per K-step per lane (1.0 loads/WMMA).
// M must be a multiple of 128, N of 256 (workspace padded accordingly).
// ---------------------------------------------------------------------------
template <bool F32OUT, bool TRANS>
__global__ __launch_bounds__(256)
void gemm_kernel(const __bf16* __restrict__ A, int lda,
                 const __bf16* __restrict__ W, int ldw,
                 const float* __restrict__ bias, int nbias,
                 __bf16* __restrict__ Cb, float* __restrict__ Cf, int ldc,
                 __bf16* __restrict__ Tb, int mgroup, size_t tbatch, int ldt,
                 int K) {
  int wid  = threadIdx.x >> 5;
  int lane = threadIdx.x & 31;
  int wr = wid >> 2, wc = wid & 3;
  int Mb = blockIdx.y * 128 + wr * 64;
  int Nb = blockIdx.x * 256 + wc * 64;

  v8f acc[4][4] = {};
  for (int k0 = 0; k0 < K; k0 += 32) {
    v16bf a[4], b[4];
#pragma unroll
    for (int t = 0; t < 4; ++t) a[t] = load_frag_bf16(A, lda, Mb + 16 * t, k0);
#pragma unroll
    for (int t = 0; t < 4; ++t) b[t] = load_frag_bf16(W, ldw, Nb + 16 * t, k0);
#pragma unroll
    for (int tm = 0; tm < 4; ++tm)
#pragma unroll
      for (int tn = 0; tn < 4; ++tn)
        acc[tm][tn] = WMMA_BF16(a[tm], b[tn], acc[tm][tn]);
  }

  int half = lane >> 4, nl = lane & 15;
#pragma unroll
  for (int tm = 0; tm < 4; ++tm) {
#pragma unroll
    for (int tn = 0; tn < 4; ++tn) {
      int nG = Nb + tn * 16 + nl;
      float bv = (bias != nullptr && nG < nbias) ? bias[nG] : 0.0f;
#pragma unroll
      for (int i = 0; i < 8; ++i) {
        int mG = Mb + tm * 16 + i + 8 * half;
        float v = acc[tm][tn][i] + bv;
        if constexpr (F32OUT) {
          Cf[(size_t)mG * ldc + nG] = v;
        } else {
          Cb[(size_t)mG * ldc + nG] = (__bf16)v;
        }
        if constexpr (TRANS) {
          // Tb layout: [m/mgroup batch][nG][m%mgroup]
          Tb[(size_t)(mG / mgroup) * tbatch + (size_t)nG * ldt + (mG % mgroup)] =
              (__bf16)v;
        }
      }
    }
  }
}

// ---------------------------------------------------------------------------
// hqw[(b*8+h)*16+q][c] = hq[(b*16+q)][c] * h2att_w[h][c]
// ---------------------------------------------------------------------------
__global__ __launch_bounds__(256)
void hqw_kernel(const __bf16* __restrict__ hq, const float* __restrict__ w,
                __bf16* __restrict__ hqw) {
  size_t idx = (size_t)blockIdx.x * 256 + threadIdx.x;  // 16,777,216 total
  int c = (int)(idx & 1023);
  int q = (int)((idx >> 10) & 15);
  int h = (int)((idx >> 14) & 7);
  int b = (int)(idx >> 17);
  float v = (float)hq[((size_t)(b * 16 + q)) * CDIM + c] * w[h * CDIM + c];
  hqw[idx] = (__bf16)v;
}

// ---------------------------------------------------------------------------
// Per (b,h): logits[q,s] = sum_c hqw[q,c]*hs[s,c] + bias[h]; joint softmax
// over all 16x256 logits; att written as bf16. The shared 16x1024 A operand
// is staged into LDS once via async global->LDS copies.
// ---------------------------------------------------------------------------
__global__ __launch_bounds__(256)
void attn_kernel(const __bf16* __restrict__ hqw, const __bf16* __restrict__ hs,
                 const float* __restrict__ h2att_b, __bf16* __restrict__ att) {
  __shared__ __bf16 Alds[NQ * CDIM];   // 32 KB
  __shared__ float  lds[NQ * NS];      // 16 KB
  __shared__ float  red[256];
  int bh = blockIdx.x;
  int b = bh >> 3, h = bh & 7;
  const __bf16* A  = hqw + (size_t)bh * NQ * CDIM;
  const __bf16* Wm = hs  + (size_t)b  * NS * CDIM;

  // Stage A: 16*1024*2 bytes = 2048 x 16B transfers, tracked by ASYNCcnt.
  for (int i = threadIdx.x; i < (NQ * CDIM) / 8; i += 256)
    async_copy_b128(&Alds[i * 8], A + (size_t)i * 8);
  wait_asynccnt0();
  __syncthreads();

  int wid = threadIdx.x >> 5, lane = threadIdx.x & 31;
  int Nb = wid * 32;
  v8f acc[2] = {};
  for (int k0 = 0; k0 < CDIM; k0 += 32) {
    v16bf a  = load_frag_bf16(Alds, CDIM, 0, k0);       // ds_load_b128
    v16bf b0 = load_frag_bf16(Wm, CDIM, Nb,      k0);
    v16bf b1 = load_frag_bf16(Wm, CDIM, Nb + 16, k0);
    acc[0] = WMMA_BF16(a, b0, acc[0]);
    acc[1] = WMMA_BF16(a, b1, acc[1]);
  }
  float bias = h2att_b[h];
  int half = lane >> 4, nl = lane & 15;
  for (int tn = 0; tn < 2; ++tn)
    for (int i = 0; i < 8; ++i) {
      int m = i + 8 * half;
      int n = Nb + tn * 16 + nl;
      lds[m * NS + n] = acc[tn][i] + bias;
    }
  __syncthreads();

  int t = threadIdx.x;
  float mx = -1e30f;
  for (int i = t; i < NQ * NS; i += 256) mx = fmaxf(mx, lds[i]);
  red[t] = mx; __syncthreads();
  for (int s = 128; s > 0; s >>= 1) {
    if (t < s) red[t] = fmaxf(red[t], red[t + s]);
    __syncthreads();
  }
  float gmax = red[0]; __syncthreads();
  float sm = 0.0f;
  for (int i = t; i < NQ * NS; i += 256) sm += __expf(lds[i] - gmax);
  red[t] = sm; __syncthreads();
  for (int s = 128; s > 0; s >>= 1) {
    if (t < s) red[t] += red[t + s];
    __syncthreads();
  }
  float inv = 1.0f / red[0];
  __bf16* ap = att + (size_t)bh * (NQ * NS);
  for (int i = t; i < NQ * NS; i += 256)
    ap[i] = (__bf16)(__expf(lds[i] - gmax) * inv);
}

// ---------------------------------------------------------------------------
// Per (b,h): t[q,c] = sum_s att[q,s]*hsT[c,s]; pooled[b][h*C+c] = sum_q hq[q,c]*t[q,c]
// att (16x256) staged into LDS via async copies; hsT streamed from global.
// ---------------------------------------------------------------------------
__global__ __launch_bounds__(256)
void pooled_kernel(const __bf16* __restrict__ att, const __bf16* __restrict__ hsT,
                   const __bf16* __restrict__ hq, __bf16* __restrict__ pooled) {
  __shared__ __bf16 Alds[NQ * NS];  // 8 KB
  int bh = blockIdx.x;
  int b = bh >> 3, h = bh & 7;
  const __bf16* A   = att + (size_t)bh * (NQ * NS);   // [16][256]
  const __bf16* Wm  = hsT + (size_t)b * CDIM * NS;    // [1024][256]
  const __bf16* hqb = hq  + (size_t)b * NQ * CDIM;    // [16][1024]

  for (int i = threadIdx.x; i < (NQ * NS) / 8; i += 256)
    async_copy_b128(&Alds[i * 8], A + (size_t)i * 8);
  wait_asynccnt0();
  __syncthreads();

  int wid = threadIdx.x >> 5, lane = threadIdx.x & 31;
  int Nb = wid * 128;  // 8 waves cover N = 1024
  v8f acc[8] = {};
  for (int k0 = 0; k0 < NS; k0 += 32) {
    v16bf a = load_frag_bf16(Alds, NS, 0, k0);          // ds_load_b128
    for (int tn = 0; tn < 8; ++tn) {
      v16bf bf = load_frag_bf16(Wm, NS, Nb + tn * 16, k0);
      acc[tn] = WMMA_BF16(a, bf, acc[tn]);
    }
  }
  int half = lane >> 4, nl = lane & 15;
  for (int tn = 0; tn < 8; ++tn) {
    int c = Nb + tn * 16 + nl;
    float part = 0.0f;
    for (int i = 0; i < 8; ++i) {
      int m = i + 8 * half;
      part += acc[tn][i] * (float)hqb[(size_t)m * CDIM + c];
    }
    part += __shfl_xor(part, 16, 32);   // combine M halves across lanes
    if (half == 0)
      pooled[(size_t)b * FCK + h * CDIM + c] = (__bf16)part;
  }
}

// ---------------------------------------------------------------------------
// log_softmax over n valid columns of sim [rows][ldsim]
// ---------------------------------------------------------------------------
__global__ __launch_bounds__(256)
void logsoftmax_kernel(const float* __restrict__ sim, int ldsim, int n,
                       float* __restrict__ out) {
  __shared__ float red[256];
  int row = blockIdx.x, t = threadIdx.x;
  const float* s = sim + (size_t)row * ldsim;
  float mx = -1e30f;
  for (int k = t; k < n; k += 256) mx = fmaxf(mx, s[k]);
  red[t] = mx; __syncthreads();
  for (int st = 128; st > 0; st >>= 1) {
    if (t < st) red[t] = fmaxf(red[t], red[t + st]);
    __syncthreads();
  }
  float gmax = red[0]; __syncthreads();
  float sm = 0.0f;
  for (int k = t; k < n; k += 256) sm += __expf(s[k] - gmax);
  red[t] = sm; __syncthreads();
  for (int st = 128; st > 0; st >>= 1) {
    if (t < st) red[t] += red[t + st];
    __syncthreads();
  }
  float lse = gmax + __logf(red[0]);
  for (int k = t; k < n; k += 256) out[(size_t)row * n + k] = s[k] - lse;
}

// ---------------------------------------------------------------------------
// Host-side orchestration
// ---------------------------------------------------------------------------
extern "C" void kernel_launch(void* const* d_in, const int* in_sizes, int n_in,
                              void* d_out, int out_size, void* d_ws, size_t ws_size,
                              hipStream_t stream) {
  (void)in_sizes; (void)n_in; (void)out_size; (void)ws_size;
  const int*   he_ques = (const int*)  d_in[0];
  const int*   he_kg   = (const int*)  d_in[1];
  const float* emb     = (const float*)d_in[2];
  const float* q2h_w   = (const float*)d_in[3];
  const float* q2h_b   = (const float*)d_in[4];
  const float* k2h_w   = (const float*)d_in[5];
  const float* k2h_b   = (const float*)d_in[6];
  const float* h2att_w = (const float*)d_in[7];
  const float* h2att_b = (const float*)d_in[8];
  const float* fc_w    = (const float*)d_in[9];
  const float* fc_b    = (const float*)d_in[10];
  const float* glove   = (const float*)d_in[11];

  size_t off = 0;
  auto alloc = [&](size_t bytes) -> void* {
    off = (off + 255) & ~(size_t)255;
    void* p = (char*)d_ws + off;
    off += bytes;
    return p;
  };

  __bf16* Aq     = (__bf16*)alloc((size_t)BB * NQ * KPACK * 2);       // 2048 x 928
  __bf16* As     = (__bf16*)alloc((size_t)BB * NS * KPACK * 2);       // 32768 x 928
  __bf16* Wqb    = (__bf16*)alloc((size_t)CDIM * KPACK * 2);          // 1024 x 928
  __bf16* Wkb    = (__bf16*)alloc((size_t)CDIM * KPACK * 2);
  __bf16* hqb    = (__bf16*)alloc((size_t)BB * NQ * CDIM * 2);        // 2048 x 1024
  __bf16* hsb    = (__bf16*)alloc((size_t)BB * NS * CDIM * 2);        // 32768 x 1024
  __bf16* hsT    = (__bf16*)alloc((size_t)BB * CDIM * NS * 2);        // [b][c][s]
  __bf16* hqw    = (__bf16*)alloc((size_t)BB * HH * NQ * CDIM * 2);   // 16384 x 1024
  __bf16* attb   = (__bf16*)alloc((size_t)BB * HH * NQ * NS * 2);     // [bh][16][256]
  __bf16* pooled = (__bf16*)alloc((size_t)BB * FCK * 2);              // 128 x 8192
  __bf16* fcwb   = (__bf16*)alloc((size_t)FCNP * FCK * 2);            // 512 x 8192
  __bf16* glvb   = (__bf16*)alloc((size_t)NAPAD * NPADO * 2);         // 5120 x 320
  __bf16* outA   = (__bf16*)alloc((size_t)BB * FCNP * 2);             // 128 x 512
  float*  sim    = (float*) alloc((size_t)BB * NAPAD * 4);            // 128 x 5120

  // 1) gather/pack inputs to bf16
  pack_rows_kernel<<<BB * NQ, 256, 0, stream>>>(he_ques, emb, Aq);
  pack_rows_kernel<<<BB * NS, 256, 0, stream>>>(he_kg, emb, As);
  // 2) weights to bf16 (zero-padded)
  cvt_pad_kernel<<<CDIM, 256, 0, stream>>>(q2h_w, CDIM, EDIM * 3, Wqb, KPACK);
  cvt_pad_kernel<<<CDIM, 256, 0, stream>>>(k2h_w, CDIM, EDIM * 3, Wkb, KPACK);
  cvt_pad_kernel<<<FCNP, 256, 0, stream>>>(fc_w, NOUT, FCK, fcwb, FCK);
  cvt_pad_kernel<<<NAPAD, 256, 0, stream>>>(glove, NANS, NOUT, glvb, NPADO);

  // 3) hq projection: [2048 x 928] * [1024 x 928]^T -> bf16 [2048 x 1024]
  gemm_kernel<false, false><<<dim3(CDIM / 256, (BB * NQ) / 128), 256, 0, stream>>>(
      Aq, KPACK, Wqb, KPACK, q2h_b, CDIM, hqb, nullptr, CDIM,
      nullptr, 1, 0, 0, KPACK);

  // 4) hs projection + transposed copy -> hs [32768x1024], hsT [b][1024][256]
  gemm_kernel<false, true><<<dim3(CDIM / 256, (BB * NS) / 128), 256, 0, stream>>>(
      As, KPACK, Wkb, KPACK, k2h_b, CDIM, hsb, nullptr, CDIM,
      hsT, NS, (size_t)CDIM * NS, NS, KPACK);

  // 5) hqw = hq * h2att_w (per head)
  hqw_kernel<<<(BB * HH * NQ * CDIM) / 256, 256, 0, stream>>>(hqb, h2att_w, hqw);

  // 6) logits + joint softmax -> att
  attn_kernel<<<BB * HH, 256, 0, stream>>>(hqw, hsb, h2att_b, attb);

  // 7) t = att @ hs, pooled = sum_q hq * t
  pooled_kernel<<<BB * HH, 256, 0, stream>>>(attb, hsT, hqb, pooled);

  // 8) fc: [128 x 8192] * [512 x 8192]^T -> bf16 [128 x 512] (cols >=300 are 0)
  gemm_kernel<false, false><<<dim3(FCNP / 256, 1), 256, 0, stream>>>(
      pooled, FCK, fcwb, FCK, fc_b, NOUT, outA, nullptr, FCNP,
      nullptr, 1, 0, 0, FCK);

  // 9) sim = out @ glove^T : [128 x 320(K)] * [5120 x 320]^T -> f32 [128 x 5120]
  gemm_kernel<true, false><<<dim3(NAPAD / 256, 1), 256, 0, stream>>>(
      outA, FCNP, glvb, NPADO, nullptr, 0, nullptr, sim, NAPAD,
      nullptr, 1, 0, 0, NPADO);

  // 10) log_softmax over 5000 valid answers
  logsoftmax_kernel<<<BB, 256, 0, stream>>>(sim, NAPAD, NANS, (float*)d_out);
}
